// naiveTransformerNet_18494129177013
// MI455X (gfx1250) — matrix operand order repaired and checked by reading, not compiled, not run on hardware
//
#include <hip/hip_runtime.h>
#include <hip/hip_bf16.h>
#include <math.h>

typedef __attribute__((ext_vector_type(16))) _Float16 v16h;
typedef __attribute__((ext_vector_type(8)))  _Float16 v8h;
typedef __attribute__((ext_vector_type(8)))  float    v8f;

#define N_NODES   100000
#define N_EDGES   1600000
#define IN_CH     128
#define EDGE_DIM  8
#define HEADS     4
#define N_GRAPHS  64
#define D1        40   // HEADS*C1
#define D2        20   // HEADS*C2

static inline int ceil_div(long a, int b) { return (int)((a + b - 1) / b); }

// ---------------- utility kernels ----------------
__global__ void k_f32_to_f16(const float* __restrict__ s, _Float16* __restrict__ d, long n) {
  long i = (long)blockIdx.x * blockDim.x + threadIdx.x;
  if (i < n) d[i] = (_Float16)s[i];
}
__global__ void k_fill_f32(float* p, float v, long n) {
  long i = (long)blockIdx.x * blockDim.x + threadIdx.x;
  if (i < n) p[i] = v;
}
__global__ void k_fill_u32(unsigned* p, unsigned v, long n) {
  long i = (long)blockIdx.x * blockDim.x + threadIdx.x;
  if (i < n) p[i] = v;
}

// Pack fused [Kpad x 4C] weights (cols = [Wq|Wk|Wv|Ws]) DIRECTLY into WMMA B-fragment
// order: frag[nt][ks][lane][j], j = 0..15 halves, where
//   k = ks*32 + (lane>>4)*8 + j (j<8), ks*32 + 16 + (lane>>4)*8 + (j-8) (j>=8)
//   n = nt*16 + (lane&15)
// so the GEMM loads each fragment with two coalesced b128 loads per lane.
__global__ void k_pack_w_frag(const float* __restrict__ Wq, const float* __restrict__ Wk,
                              const float* __restrict__ Wv, const float* __restrict__ Ws,
                              _Float16* __restrict__ out, int Kin, int Ksteps, int C, int Ntiles) {
  int i = blockIdx.x * blockDim.x + threadIdx.x;
  int total = Ntiles * Ksteps * 32 * 16;
  if (i >= total) return;
  int j = i & 15;
  int lane = (i >> 4) & 31;
  int rest = i >> 9;
  int ks = rest % Ksteps;
  int nt = rest / Ksteps;
  int k = ks * 32 + (lane >> 4) * 8 + (j < 8 ? j : 16 + (j - 8));
  int n = nt * 16 + (lane & 15);
  float v = 0.f;
  if (k < Kin) {
    int m = n / C, cc = n - m * C;
    const float* W = (m == 0) ? Wq : (m == 1) ? Wk : (m == 2) ? Wv : Ws;
    v = W[k * C + cc];
  }
  out[i] = (_Float16)v;
}
__global__ void k_pack_b(const float* __restrict__ bq, const float* __restrict__ bk,
                         const float* __restrict__ bv, const float* __restrict__ bs,
                         float* __restrict__ out, int C) {
  int i = blockIdx.x * blockDim.x + threadIdx.x;
  if (i >= 4 * C) return;
  int m = i / C, c = i - m * C;
  const float* b = (m == 0) ? bq : (m == 1) ? bk : (m == 2) ? bv : bs;
  out[i] = b[c];
}

// ---------------- WMMA GEMM: C[M x 16*NT] = A_f16[M x 32*KSTEPS] @ Bfrag + bias ----------
// One wave per 16-row M-tile; all NT output tiles held in registers (A reused across N).
template <int NT, int KSTEPS>
__global__ void k_gemm_wmma(const _Float16* __restrict__ A, int lda,
                            const _Float16* __restrict__ Bfrag,
                            const float* __restrict__ bias,
                            float* __restrict__ C, int ldc, int Mtiles) {
  int wave = threadIdx.x >> 5;
  int lane = threadIdx.x & 31;
  int mt = blockIdx.x * (blockDim.x >> 5) + wave;
  if (mt >= Mtiles) return;                      // wave-uniform: EXEC stays all-ones for WMMA
  int half16 = lane >> 4;                        // 0: lanes 0-15, 1: lanes 16-31
  int l15 = lane & 15;
  const _Float16* Ap = A + (long)(mt * 16 + l15) * lda + half16 * 8;
  v8f zero = {};
  v8f acc[NT];
#pragma unroll
  for (int nt = 0; nt < NT; ++nt) acc[nt] = zero;
#pragma unroll
  for (int ks = 0; ks < KSTEPS; ++ks) {
    v8h alo = *(const v8h*)(Ap + ks * 32);       // A: K = {0..7 | 8..15} of this k-step
    v8h ahi = *(const v8h*)(Ap + ks * 32 + 16);  //    K = {16..23 | 24..31}
    v16h a;
#pragma unroll
    for (int i = 0; i < 8; ++i) { a[i] = alo[i]; a[8 + i] = ahi[i]; }
#pragma unroll
    for (int nt = 0; nt < NT; ++nt) {
      const _Float16* Bp = Bfrag + (((nt * KSTEPS + ks) * 32 + lane) << 4);
      v8h blo = *(const v8h*)(Bp);
      v8h bhi = *(const v8h*)(Bp + 8);
      v16h b;
#pragma unroll
      for (int i = 0; i < 8; ++i) { b[i] = blo[i]; b[8 + i] = bhi[i]; }
      acc[nt] = __builtin_amdgcn_wmma_f32_16x16x32_f16(false, a, false, b, (short)0,
                                                       acc[nt], false, false);
    }
  }
  int mbase = mt * 16 + half16 * 8;              // D layout: VGPR i -> M = i (+8 for lanes 16-31)
#pragma unroll
  for (int nt = 0; nt < NT; ++nt) {
    int ncol = nt * 16 + l15;
    float bn = bias[ncol];
#pragma unroll
    for (int i = 0; i < 8; ++i)
      C[(long)(mbase + i) * ldc + ncol] = acc[nt][i] + bn;
  }
}

// ---------------- order-preserving float<->uint key for atomicMax ----------------
__device__ __forceinline__ unsigned f32key(float f) {
  unsigned u = __float_as_uint(f);
  return (u & 0x80000000u) ? ~u : (u | 0x80000000u);
}
__device__ __forceinline__ float keyf32(unsigned k) {
  return (k & 0x80000000u) ? __uint_as_float(k & 0x7fffffffu) : __uint_as_float(~k);
}

// ---------------- edge pass A: attention logits + segment max ----------------
template <int CH>
__global__ void k_edge_alpha(const int* __restrict__ srcI, const int* __restrict__ dstI,
                             const float* __restrict__ eattr, const float* __restrict__ We,
                             const float* __restrict__ QKVS, int ld, int kOff,
                             float* __restrict__ alpha, unsigned* __restrict__ nodeMaxK,
                             float invSqrtC) {
  __shared__ float sWe[EDGE_DIM * HEADS * CH];
  for (int i = threadIdx.x; i < EDGE_DIM * HEADS * CH; i += blockDim.x) sWe[i] = We[i];
  __syncthreads();
  long idx = (long)blockIdx.x * blockDim.x + threadIdx.x;
  if (idx >= (long)N_EDGES * HEADS) return;
  int e = (int)(idx >> 2), h = (int)(idx & 3);
  int s = srcI[e], d = dstI[e];
  float ea[EDGE_DIM];
#pragma unroll
  for (int i = 0; i < EDGE_DIM; ++i) ea[i] = eattr[(long)e * EDGE_DIM + i];
  const float* q = QKVS + (long)d * ld + h * CH;
  const float* kk = QKVS + (long)s * ld + kOff + h * CH;
  float a = 0.f;
#pragma unroll
  for (int c = 0; c < CH; ++c) {
    float ec = 0.f;
#pragma unroll
    for (int i = 0; i < EDGE_DIM; ++i) ec += ea[i] * sWe[i * HEADS * CH + h * CH + c];
    a += q[c] * (kk[c] + ec);
  }
  a *= invSqrtC;
  alpha[idx] = a;
  atomicMax(&nodeMaxK[d * HEADS + h], f32key(a));
}

// ---------------- edge pass B: exp, denominator, unnormalized message scatter ----------------
template <int CH>
__global__ void k_edge_scatter(const int* __restrict__ srcI, const int* __restrict__ dstI,
                               const float* __restrict__ eattr, const float* __restrict__ We,
                               const float* __restrict__ QKVS, int ld, int vOff,
                               const float* __restrict__ alpha,
                               const unsigned* __restrict__ nodeMaxK,
                               float* __restrict__ den, float* __restrict__ agg) {
  __shared__ float sWe[EDGE_DIM * HEADS * CH];
  for (int i = threadIdx.x; i < EDGE_DIM * HEADS * CH; i += blockDim.x) sWe[i] = We[i];
  __syncthreads();
  long idx = (long)blockIdx.x * blockDim.x + threadIdx.x;
  if (idx >= (long)N_EDGES * HEADS) return;
  int e = (int)(idx >> 2), h = (int)(idx & 3);
  int s = srcI[e], d = dstI[e];
  float m = keyf32(nodeMaxK[d * HEADS + h]);
  float ex = __expf(alpha[idx] - m);
  atomicAdd(&den[d * HEADS + h], ex);
  float ea[EDGE_DIM];
#pragma unroll
  for (int i = 0; i < EDGE_DIM; ++i) ea[i] = eattr[(long)e * EDGE_DIM + i];
  const float* v = QKVS + (long)s * ld + vOff + h * CH;
  float* o = agg + (long)d * (HEADS * CH) + h * CH;
#pragma unroll
  for (int c = 0; c < CH; ++c) {
    float ec = 0.f;
#pragma unroll
    for (int i = 0; i < EDGE_DIM; ++i) ec += ea[i] * sWe[i * HEADS * CH + h * CH + c];
    atomicAdd(&o[c], (v[c] + ec) * ex);
  }
}

// ---------------- conv1 finalize: normalize + skip + relu, emit f16 padded to K=64 ----------------
__global__ void k_final1(const float* __restrict__ agg, const float* __restrict__ den,
                         const float* __restrict__ QKVS, _Float16* __restrict__ H1h) {
  long idx = (long)blockIdx.x * blockDim.x + threadIdx.x;
  if (idx >= (long)N_NODES * 64) return;
  int n = (int)(idx >> 6), c = (int)(idx & 63);
  float v = 0.f;
  if (c < D1) {
    int h = c / 10;
    v = agg[(long)n * D1 + c] / (den[n * HEADS + h] + 1e-16f) + QKVS[(long)n * 160 + 120 + c];
    v = fmaxf(v, 0.f);
  }
  H1h[idx] = (_Float16)v;
}

// ---------------- conv2 finalize + global mean-pool accumulation ----------------
__global__ void k_final2_pool(const float* __restrict__ agg, const float* __restrict__ den,
                              const float* __restrict__ QKVS2, const int* __restrict__ batch,
                              float* __restrict__ pool, float* __restrict__ cnt) {
  long idx = (long)blockIdx.x * blockDim.x + threadIdx.x;
  if (idx >= (long)N_NODES * D2) return;
  int n = (int)(idx / D2), c = (int)(idx - (long)n * D2);
  int h = c / 5;
  float v = agg[idx] / (den[n * HEADS + h] + 1e-16f) + QKVS2[(long)n * 80 + 60 + c];
  v = fmaxf(v, 0.f);
  int g = batch[n];
  atomicAdd(&pool[g * D2 + c], v);
  if (c == 0) atomicAdd(&cnt[g], 1.0f);
}

// ---------------- MLP head: pooled -> 10 -> 3 ----------------
__global__ void k_head(const float* __restrict__ pool, const float* __restrict__ cnt,
                       const float* __restrict__ W1, const float* __restrict__ b1,
                       const float* __restrict__ W2, const float* __restrict__ b2,
                       float* __restrict__ out) {
  int g = threadIdx.x;
  if (g >= N_GRAPHS) return;
  float inv = 1.f / fmaxf(cnt[g], 1.f);
  float p[D2];
#pragma unroll
  for (int c = 0; c < D2; ++c) p[c] = pool[g * D2 + c] * inv;
  float hd[10];
#pragma unroll
  for (int j = 0; j < 10; ++j) {
    float s = b1[j];
#pragma unroll
    for (int c = 0; c < D2; ++c) s += p[c] * W1[c * 10 + j];
    hd[j] = fmaxf(s, 0.f);
  }
#pragma unroll
  for (int t = 0; t < 3; ++t) {
    float s = b2[t];
#pragma unroll
    for (int j = 0; j < 10; ++j) s += hd[j] * W2[j * 3 + t];
    out[g * 3 + t] = s;
  }
}

extern "C" void kernel_launch(void* const* d_in, const int* in_sizes, int n_in,
                              void* d_out, int out_size, void* d_ws, size_t ws_size,
                              hipStream_t stream) {
  const float* x     = (const float*)d_in[0];
  const int*   eidx  = (const int*)d_in[1];
  const int*   srcI  = eidx;
  const int*   dstI  = eidx + N_EDGES;
  const float* eattr = (const float*)d_in[2];
  const int*   batch = (const int*)d_in[3];
  const float *Wq1 = (const float*)d_in[4],  *bq1 = (const float*)d_in[5];
  const float *Wk1 = (const float*)d_in[6],  *bk1 = (const float*)d_in[7];
  const float *Wv1 = (const float*)d_in[8],  *bv1 = (const float*)d_in[9];
  const float *We1 = (const float*)d_in[10];
  const float *Ws1 = (const float*)d_in[11], *bs1 = (const float*)d_in[12];
  const float *Wq2 = (const float*)d_in[13], *bq2 = (const float*)d_in[14];
  const float *Wk2 = (const float*)d_in[15], *bk2 = (const float*)d_in[16];
  const float *Wv2 = (const float*)d_in[17], *bv2 = (const float*)d_in[18];
  const float *We2 = (const float*)d_in[19];
  const float *Ws2 = (const float*)d_in[20], *bs2 = (const float*)d_in[21];
  const float *W1  = (const float*)d_in[22], *b1  = (const float*)d_in[23];
  const float *W2  = (const float*)d_in[24], *b2  = (const float*)d_in[25];
  float* out = (float*)d_out;

  // ---- workspace layout (256B aligned slabs) ----
  char* ws = (char*)d_ws;
  auto alloc = [&](size_t bytes) -> char* {
    char* p = ws; ws += (bytes + 255) & ~(size_t)255; return p;
  };
  _Float16* Xh     = (_Float16*)alloc((size_t)N_NODES * IN_CH * 2);    // x in f16
  _Float16* W1pk   = (_Float16*)alloc((size_t)10 * 4 * 32 * 16 * 2);   // conv1 B fragments
  _Float16* W2pk   = (_Float16*)alloc((size_t)5 * 2 * 32 * 16 * 2);    // conv2 B fragments
  float*    B1pk   = (float*)alloc(160 * 4);
  float*    B2pk   = (float*)alloc(80 * 4);
  float*    QKVS1  = (float*)alloc((size_t)N_NODES * 160 * 4);         // [q|k|v|s]
  float*    QKVS2  = (float*)alloc((size_t)N_NODES * 80 * 4);
  float*    alphaB = (float*)alloc((size_t)N_EDGES * HEADS * 4);       // reused conv1/conv2
  unsigned* maxK   = (unsigned*)alloc((size_t)N_NODES * HEADS * 4);
  float*    den    = (float*)alloc((size_t)N_NODES * HEADS * 4);
  float*    agg1   = (float*)alloc((size_t)N_NODES * D1 * 4);
  _Float16* H1h    = (_Float16*)alloc((size_t)N_NODES * 64 * 2);       // relu(conv1), K-padded
  float*    agg2   = (float*)alloc((size_t)N_NODES * D2 * 4);
  float*    pool   = (float*)alloc((size_t)N_GRAPHS * D2 * 4);
  float*    cnt    = (float*)alloc((size_t)N_GRAPHS * 4);

  const int B = 256;
  const long nEH = (long)N_EDGES * HEADS;

  // ---- stage inputs: x -> f16, pack fragment-order weight/bias blocks ----
  k_f32_to_f16<<<ceil_div((long)N_NODES * IN_CH, B), B, 0, stream>>>(x, Xh, (long)N_NODES * IN_CH);
  k_pack_w_frag<<<ceil_div(10 * 4 * 512, B), B, 0, stream>>>(Wq1, Wk1, Wv1, Ws1, W1pk,
                                                             IN_CH, 4, D1, 10);
  k_pack_w_frag<<<ceil_div(5 * 2 * 512, B), B, 0, stream>>>(Wq2, Wk2, Wv2, Ws2, W2pk,
                                                            D1, 2, D2, 5);
  k_pack_b<<<1, B, 0, stream>>>(bq1, bk1, bv1, bs1, B1pk, D1);
  k_pack_b<<<1, B, 0, stream>>>(bq2, bk2, bv2, bs2, B2pk, D2);

  // ---- conv1: fused QKVS GEMM (100000x128 @ 128x160), WMMA f16->f32 ----
  k_gemm_wmma<10, 4><<<ceil_div(N_NODES / 16, 8), 256, 0, stream>>>(
      Xh, IN_CH, W1pk, B1pk, QKVS1, 160, N_NODES / 16);

  // ---- conv1 attention ----
  k_fill_u32<<<ceil_div((long)N_NODES * HEADS, B), B, 0, stream>>>(maxK, 0u, (long)N_NODES * HEADS);
  k_fill_f32<<<ceil_div((long)N_NODES * HEADS, B), B, 0, stream>>>(den, 0.f, (long)N_NODES * HEADS);
  k_fill_f32<<<ceil_div((long)N_NODES * D1, B), B, 0, stream>>>(agg1, 0.f, (long)N_NODES * D1);
  k_edge_alpha<10><<<ceil_div(nEH, B), B, 0, stream>>>(
      srcI, dstI, eattr, We1, QKVS1, 160, D1, alphaB, maxK, 1.0f / sqrtf(10.0f));
  k_edge_scatter<10><<<ceil_div(nEH, B), B, 0, stream>>>(
      srcI, dstI, eattr, We1, QKVS1, 160, 2 * D1, alphaB, maxK, den, agg1);
  k_final1<<<ceil_div((long)N_NODES * 64, B), B, 0, stream>>>(agg1, den, QKVS1, H1h);

  // ---- conv2: fused QKVS GEMM (100000x64pad @ 64x80) ----
  k_gemm_wmma<5, 2><<<ceil_div(N_NODES / 16, 8), 256, 0, stream>>>(
      H1h, 64, W2pk, B2pk, QKVS2, 80, N_NODES / 16);

  // ---- conv2 attention ----
  k_fill_u32<<<ceil_div((long)N_NODES * HEADS, B), B, 0, stream>>>(maxK, 0u, (long)N_NODES * HEADS);
  k_fill_f32<<<ceil_div((long)N_NODES * HEADS, B), B, 0, stream>>>(den, 0.f, (long)N_NODES * HEADS);
  k_fill_f32<<<ceil_div((long)N_NODES * D2, B), B, 0, stream>>>(agg2, 0.f, (long)N_NODES * D2);
  k_edge_alpha<5><<<ceil_div(nEH, B), B, 0, stream>>>(
      srcI, dstI, eattr, We2, QKVS2, 80, D2, alphaB, maxK, 1.0f / sqrtf(5.0f));
  k_edge_scatter<5><<<ceil_div(nEH, B), B, 0, stream>>>(
      srcI, dstI, eattr, We2, QKVS2, 80, 2 * D2, alphaB, maxK, den, agg2);

  // ---- pooling + head ----
  k_fill_f32<<<1, B, 0, stream>>>(pool, 0.f, (long)N_GRAPHS * D2);
  k_fill_f32<<<1, B, 0, stream>>>(cnt, 0.f, (long)N_GRAPHS);
  k_final2_pool<<<ceil_div((long)N_NODES * D2, B), B, 0, stream>>>(agg2, den, QKVS2, batch, pool, cnt);
  k_head<<<1, 64, 0, stream>>>(pool, cnt, W1, b1, W2, b2, out);
}
